// Model_635655160220
// MI455X (gfx1250) — matrix operand order repaired and checked by reading, not compiled, not run on hardware
//
#include <hip/hip_runtime.h>
#include <hip/hip_bf16.h>
#include <math.h>

typedef __attribute__((ext_vector_type(16))) _Float16 v16h;
typedef __attribute__((ext_vector_type(8)))  _Float16 v8h;
typedef __attribute__((ext_vector_type(8)))  float    v8f;

#define NB    2
#define NWAY  5
#define NQ    16
#define NS    25
#define CCH   640
#define HW    25
#define FSZ   (CCH*HW)      // 16000
#define PTOT  (NB*NQ*NWAY)  // 160
#define KDIM  16000         // GEMM K (tap-major, channel-minor)
#define NCG   20            // deform channel chunks (32 ch each)

// ---------------- prototype: proto[b,k,f] = sum_n st[b,n,k]*feat[b,n,f] / sum_n st ----------------
__global__ void proto_kernel(const float* __restrict__ sf, const float* __restrict__ st,
                             float* __restrict__ proto) {
    int idx = blockIdx.x * blockDim.x + threadIdx.x;   // NB*NWAY*FSZ = 160000
    if (idx >= NB * NWAY * FSZ) return;
    int f   = idx % FSZ;
    int k   = (idx / FSZ) % NWAY;
    int b_i = idx / (NWAY * FSZ);
    float num = 0.f, den = 0.f;
    for (int n = 0; n < NS; ++n) {
        float w = st[(b_i * NS + n) * NWAY + k];
        num += w * sf[(size_t)(b_i * NS + n) * FSZ + f];
        den += w;
    }
    proto[idx] = num / den;
}

// ---------------- per-(b,k,h,w) channel norm of proto ----------------
__global__ void nrm_kernel(const float* __restrict__ proto, float* __restrict__ nrm) {
    int idx = blockIdx.x * blockDim.x + threadIdx.x;   // NB*NWAY*HW = 250
    if (idx >= NB * NWAY * HW) return;
    int s = idx % HW, bk = idx / HW;
    float acc = 0.f;
    for (int c = 0; c < CCH; ++c) {
        float v = proto[(size_t)bk * FSZ + c * HW + s];
        acc += v * v;
    }
    nrm[idx] = fmaxf(sqrtf(acc), 1e-12f);
}

// ---------------- cw[b,k,c] = mean_hw proto/nrm ----------------
__global__ void cw_kernel(const float* __restrict__ proto, const float* __restrict__ nrm,
                          float* __restrict__ cw) {
    int idx = blockIdx.x * blockDim.x + threadIdx.x;   // NB*NWAY*CCH = 6400
    if (idx >= NB * NWAY * CCH) return;
    int c = idx % CCH, bk = idx / CCH;
    float acc = 0.f;
    for (int s = 0; s < HW; ++s)
        acc += proto[(size_t)bk * FSZ + c * HW + s] / nrm[bk * HW + s];
    cw[idx] = acc * (1.0f / HW);
}

// ---------------- reorder + f16-convert offset-conv weights: k = tap*640 + c ----------------
__global__ void prep_w(const float* __restrict__ w_off, _Float16* __restrict__ w_q,
                       _Float16* __restrict__ w_k) {
    int idx = blockIdx.x * blockDim.x + threadIdx.x;   // 18*16000 = 288000
    if (idx >= 18 * KDIM) return;
    int o = idx / KDIM;
    int k = idx % KDIM;
    int t = k / CCH, c = k % CCH;
    w_q[idx] = (_Float16)w_off[(size_t)(o * 1280 + c) * 25 + t];
    w_k[idx] = (_Float16)w_off[(size_t)(o * 1280 + 640 + c) * 25 + t];
}

// ---------------- WMMA GEMM: out[18,npos] = Wh[18,16000] x im2col(src)[16000,npos] ----------------
__global__ void __launch_bounds__(32)
conv_gemm(const _Float16* __restrict__ Wh, const float* __restrict__ src,
          float* __restrict__ out, int npos) {
    int lane = threadIdx.x;
    int col  = blockIdx.x * 16 + (lane & 15);
    bool colValid = (col < npos);
    int cc  = colValid ? col : 0;
    int map = cc / HW;
    int s   = cc % HW;
    int y = s / 5, x = s % 5;
    const float* base = src + (size_t)map * FSZ;

    int m0     = lane & 15;
    int kshift = (lane >> 4) * 16;
    int kB16   = (lane >> 4) * 16;

    v8f acc0 = {}; v8f acc1 = {};

    for (int kb = 0; kb < KDIM; kb += 32) {
        // A tile 0 (rows 0..15): lanes 0-15 K{0-15}, lanes 16-31 K{16-31}
        const _Float16* w0 = Wh + (size_t)m0 * KDIM + kb + kshift;
        __builtin_prefetch(w0 + 64, 0, 3);
        v8h lo0 = *(const v8h*)w0;
        v8h hi0 = *(const v8h*)(w0 + 8);
        v16h a0;
#pragma unroll
        for (int i = 0; i < 8; ++i) { a0[i] = lo0[i]; a0[8 + i] = hi0[i]; }

        // A tile 1 (rows 16..31; only 16,17 real)
        v16h a1;
        if (m0 < 2) {
            const _Float16* w1 = Wh + (size_t)(16 + m0) * KDIM + kb + kshift;
            v8h lo1 = *(const v8h*)w1;
            v8h hi1 = *(const v8h*)(w1 + 8);
#pragma unroll
            for (int i = 0; i < 8; ++i) { a1[i] = lo1[i]; a1[8 + i] = hi1[i]; }
        } else {
#pragma unroll
            for (int i = 0; i < 16; ++i) a1[i] = (_Float16)0.0f;
        }

        // B fragment: lane's column = col, 16 consecutive K (one conv tap, 16 channels)
        int k0 = kb + kB16;
        int t  = k0 / CCH;
        int c0 = k0 % CCH;
        int yy = y + t / 5 - 2;
        int xx = x + t % 5 - 2;
        bool v = colValid && (yy >= 0) && (yy < 5) && (xx >= 0) && (xx < 5);
        v16h bfrag;
        if (v) {
            const float* sp = base + (size_t)c0 * HW + yy * 5 + xx;
#pragma unroll
            for (int i = 0; i < 16; ++i) bfrag[i] = (_Float16)sp[(size_t)i * HW];
        } else {
#pragma unroll
            for (int i = 0; i < 16; ++i) bfrag[i] = (_Float16)0.0f;
        }

        acc0 = __builtin_amdgcn_wmma_f32_16x16x32_f16(false, a0, false, bfrag,
                                                      (short)0, acc0, false, false);
        acc1 = __builtin_amdgcn_wmma_f32_16x16x32_f16(false, a1, false, bfrag,
                                                      (short)0, acc1, false, false);
    }

    if (colValid) {
        int half8 = (lane >> 4) * 8;
#pragma unroll
        for (int i = 0; i < 8; ++i) {
            int m = i + half8;
            if (m < 18)      out[(size_t)m        * npos + col] = acc0[i];
            if (16 + m < 18) out[(size_t)(16 + m) * npos + col] = acc1[i];
        }
    }
}

// ---------------- deformable 3x3 sampling + einsum, split over channel chunks ----------------
// thread = (p, s, g): partial acc over channels [g*32, g*32+32)
__global__ void deform_part(const float* __restrict__ proto, const float* __restrict__ conv_q,
                            const float* __restrict__ conv_k, const float* __restrict__ w_def,
                            float* __restrict__ pacc) {
    int idx = blockIdx.x * blockDim.x + threadIdx.x;   // PTOT*HW*NCG = 80000
    if (idx >= PTOT * HW * NCG) return;
    int g    = idx % NCG;
    int rest = idx / NCG;
    int s = rest % HW, p = rest / HW;
    int kcls = p % NWAY, q = (p / NWAY) % NQ, b_i = p / (NWAY * NQ);
    int y = s / 5, x = s % 5;
    const float* ftr = proto + (size_t)(b_i * NWAY + kcls) * FSZ;
    int posq = (b_i * NQ + q) * HW + s;
    int posk = (b_i * NWAY + kcls) * HW + s;
    int cbeg = g * 32;

    float acc[9];
#pragma unroll
    for (int o = 0; o < 9; ++o) acc[o] = 0.f;

    for (int kk = 0; kk < 9; ++kk) {
        float offy = conv_q[(size_t)(2 * kk)     * 800 + posq] + conv_k[(size_t)(2 * kk)     * 250 + posk];
        float offx = conv_q[(size_t)(2 * kk + 1) * 800 + posq] + conv_k[(size_t)(2 * kk + 1) * 250 + posk];
        int ky = kk / 3, kx = kk % 3;
        float py = (float)(y + ky - 1) + offy;
        float px = (float)(x + kx - 1) + offx;
        float y0f = floorf(py), x0f = floorf(px);
        float ty = py - y0f, tx = px - x0f;
        int y0 = (int)y0f, x0 = (int)x0f;

        float wts[4]; int idxs[4];
        int   ys[2] = { y0, y0 + 1 }, xs[2] = { x0, x0 + 1 };
        float fy[2] = { 1.f - ty, ty }, fx[2] = { 1.f - tx, tx };
        int n = 0;
#pragma unroll
        for (int a = 0; a < 2; ++a)
#pragma unroll
            for (int bb = 0; bb < 2; ++bb) {
                bool vv = (ys[a] >= 0) && (ys[a] < 5) && (xs[bb] >= 0) && (xs[bb] < 5);
                int yc = min(max(ys[a], 0), 4), xc = min(max(xs[bb], 0), 4);
                idxs[n] = yc * 5 + xc;
                wts[n]  = vv ? fy[a] * fx[bb] : 0.f;
                ++n;
            }

        for (int c = cbeg; c < cbeg + 32; ++c) {
            const float* xc = ftr + (size_t)c * HW;
            float samp = wts[0] * xc[idxs[0]] + wts[1] * xc[idxs[1]]
                       + wts[2] * xc[idxs[2]] + wts[3] * xc[idxs[3]];
#pragma unroll
            for (int o = 0; o < 9; ++o)
                acc[o] += w_def[(size_t)(o * CCH + c) * 9 + kk] * samp;
        }
    }
#pragma unroll
    for (int o = 0; o < 9; ++o)
        pacc[((size_t)(p * HW + s) * NCG + g) * 9 + o] = acc[o];
}

// reduce chunk partials + sigmoid -> filt[p,9,h,w]
__global__ void deform_reduce(const float* __restrict__ pacc, float* __restrict__ filt) {
    int idx = blockIdx.x * blockDim.x + threadIdx.x;   // PTOT*HW*9 = 36000
    if (idx >= PTOT * HW * 9) return;
    int o = idx % 9;
    int rest = idx / 9;
    int s = rest % HW, p = rest / HW;
    float a = 0.f;
    for (int g = 0; g < NCG; ++g)
        a += pacc[((size_t)(p * HW + s) * NCG + g) * 9 + o];
    filt[(size_t)(p * 9 + o) * HW + s] = 1.0f / (1.0f + expf(-a));
}

// ---------------- RK4 dynamic filtering + classification partials ----------------
// block = 256 threads: 8 channels x 32 slots. grid = PTOT*80.
__global__ void rk_kernel(const float* __restrict__ query, const float* __restrict__ filt,
                          const float* __restrict__ cw, float* __restrict__ partial) {
    __shared__ float sfilt[225];
    __shared__ float plane[8][HW];
    __shared__ float red[256];

    int blk = blockIdx.x;
    int p = blk / 80;
    int cchunk = blk % 80;
    int kcls = p % NWAY, q = (p / NWAY) % NQ, b_i = p / (NWAY * NQ);

    int tid = threadIdx.x;
    int cr  = tid >> 5;
    int s   = tid & 31;
    int c   = cchunk * 8 + cr;
    bool act = (s < HW);
    int y = act ? s / 5 : 0, x = act ? s % 5 : 0;

    // async memory->LDS copy of this p's dynamic filter (225 floats), tracked by ASYNCcnt
    if (tid < 225) {
        unsigned lds_off = (unsigned)(unsigned long long)(void*)&sfilt[tid];
        unsigned long long gaddr = (unsigned long long)(const void*)(filt + (size_t)p * 225 + tid);
        asm volatile("global_load_async_to_lds_b32 %0, %1, off"
                     :: "v"(lds_off), "v"(gaddr) : "memory");
    }
    asm volatile("s_wait_asynccnt 0x0" ::: "memory");
    __syncthreads();

    float fte = act ? query[((size_t)(b_i * NQ + q) * CCH + c) * HW + s] : 0.f;

    auto dynconv = [&](float cur) -> float {
        if (act) plane[cr][s] = cur;
        __syncthreads();
        float r = 0.f;
        if (act) {
#pragma unroll
            for (int kk = 0; kk < 9; ++kk) {
                int yy = y + kk / 3 - 1, xx = x + kk % 3 - 1;
                float v = (yy >= 0 && yy < 5 && xx >= 0 && xx < 5) ? plane[cr][yy * 5 + xx] : 0.f;
                // reference reshapes filt to (-1,9), scrambling (kk,h,w): flat idx = s*9+kk
                r += v * sfilt[s * 9 + kk];
            }
            r = fmaxf(r, 0.f);
        }
        __syncthreads();
        return r;
    };

    float k1 = dynconv(fte);
    float k2 = dynconv(fte + k1 * (1.0f / 3.0f));
    float k3 = dynconv(fte + (k2 - k1 * (1.0f / 3.0f)));
    float k4 = dynconv(fte + (k1 - k2 + k3));
    float score = fte + (k1 + 3.0f * (k2 + k3) + k4) * 0.125f;

    float contrib = act ? score * cw[(size_t)(b_i * NWAY + kcls) * CCH + c] : 0.f;
    red[tid] = contrib;
    __syncthreads();
    for (int off = 128; off > 0; off >>= 1) {
        if (tid < off) red[tid] += red[tid + off];
        __syncthreads();
    }
    if (tid == 0) partial[(size_t)p * 80 + cchunk] = red[0];
}

__global__ void final_kernel(const float* __restrict__ partial, float* __restrict__ out) {
    int p = blockIdx.x * blockDim.x + threadIdx.x;
    if (p < PTOT) {
        float a = 0.f;
        for (int i = 0; i < 80; ++i) a += partial[(size_t)p * 80 + i];
        out[p] = a * (1.0f / 49.0f);
    }
}

extern "C" void kernel_launch(void* const* d_in, const int* in_sizes, int n_in,
                              void* d_out, int out_size, void* d_ws, size_t ws_size,
                              hipStream_t stream) {
    const float* support_feat = (const float*)d_in[0];   // (2,25,640,5,5)
    const float* query_feat   = (const float*)d_in[1];   // (2,16,640,5,5)
    const float* support_tg   = (const float*)d_in[2];   // (2,25,5)
    const float* w_off        = (const float*)d_in[4];   // (18,1280,5,5)
    const float* w_def        = (const float*)d_in[5];   // (9,640,3,3)
    float* out = (float*)d_out;                          // (32,5) flat = P

    char* wsb = (char*)d_ws;
    size_t off = 0;
    auto alloc = [&](size_t bytes) -> void* {
        void* pp = wsb + off;
        off = (off + bytes + 255) & ~(size_t)255;
        return pp;
    };
    float*    proto   = (float*)   alloc((size_t)NB * NWAY * FSZ * 4);
    float*    nrm     = (float*)   alloc((size_t)NB * NWAY * HW * 4);
    float*    cw      = (float*)   alloc((size_t)NB * NWAY * CCH * 4);
    _Float16* w_q     = (_Float16*)alloc((size_t)18 * KDIM * 2);
    _Float16* w_k     = (_Float16*)alloc((size_t)18 * KDIM * 2);
    float*    conv_q  = (float*)   alloc((size_t)18 * 800 * 4);
    float*    conv_k  = (float*)   alloc((size_t)18 * 250 * 4);
    float*    filt    = (float*)   alloc((size_t)PTOT * 9 * HW * 4);
    float*    pacc    = (float*)   alloc((size_t)PTOT * HW * NCG * 9 * 4);  // 2.88 MB
    float*    partial = (float*)   alloc((size_t)PTOT * 80 * 4);
    (void)ws_size; (void)in_sizes; (void)n_in; (void)out_size;

    proto_kernel<<<(NB * NWAY * FSZ + 255) / 256, 256, 0, stream>>>(support_feat, support_tg, proto);
    nrm_kernel<<<1, 256, 0, stream>>>(proto, nrm);
    cw_kernel<<<(NB * NWAY * CCH + 255) / 256, 256, 0, stream>>>(proto, nrm, cw);
    prep_w<<<(18 * KDIM + 255) / 256, 256, 0, stream>>>(w_off, w_q, w_k);

    conv_gemm<<<(800 + 15) / 16, 32, 0, stream>>>(w_q, query_feat, conv_q, 800);
    conv_gemm<<<(250 + 15) / 16, 32, 0, stream>>>(w_k, proto, conv_k, 250);

    deform_part<<<(PTOT * HW * NCG + 255) / 256, 256, 0, stream>>>(proto, conv_q, conv_k, w_def, pacc);
    deform_reduce<<<(PTOT * HW * 9 + 255) / 256, 256, 0, stream>>>(pacc, filt);

    rk_kernel<<<PTOT * 80, 256, 0, stream>>>(query_feat, filt, cw, partial);
    final_kernel<<<1, 256, 0, stream>>>(partial, out);
}
